// ProductionSplatFlowAttention_43224550867593
// MI455X (gfx1250) — matrix operand order repaired and checked by reading, not compiled, not run on hardware
//
#include <hip/hip_runtime.h>
#include <hip/hip_bf16.h>

// ---------------------------------------------------------------------------
// Gaussian-splat RBF attention for MI455X (gfx1250, wave32, WMMA).
// B=4, S=8192, D=512, K=20.
// GEMMs: 64-row blocks, A tile (64KB) async-staged to LDS, per-wave 64x64
// macro-tile = 4x4 v_wmma_f32_16x16x32_bf16 accumulators (4x B reuse).
// attn kernel fuses the x -> bf16 conversion (single HBM pass over x).
// ---------------------------------------------------------------------------

typedef __attribute__((ext_vector_type(16))) __bf16 v16bf;
typedef __attribute__((ext_vector_type(8)))  __bf16 bf16x8;
typedef __attribute__((ext_vector_type(2)))  __bf16 bf16x2;
typedef __attribute__((ext_vector_type(8)))  float  v8f;
typedef __attribute__((ext_vector_type(4)))  int    v4i;

#define NTOK   32768   // B*S
#define DDIM   512
#define KSPL   20
#define SEQ    8192

// ---------------------------------------------------------------------------
// Async global->LDS staging (CDNA5).
// ---------------------------------------------------------------------------
#if defined(__HIP_DEVICE_COMPILE__) &&                                        \
    __has_builtin(__builtin_amdgcn_global_load_async_to_lds_b128) &&          \
    __has_builtin(__builtin_amdgcn_s_wait_asynccnt)
#define HAVE_ASYNC_LDS 1
#else
#define HAVE_ASYNC_LDS 0
#endif

#if HAVE_ASYNC_LDS
typedef __attribute__((address_space(1))) v4i gv4i_t;
typedef __attribute__((address_space(3))) v4i lv4i_t;
#endif

__device__ __forceinline__ void async_copy_b128(void* lds_dst, const void* gsrc) {
#if HAVE_ASYNC_LDS
    __builtin_amdgcn_global_load_async_to_lds_b128(
        (gv4i_t*)(gsrc), (lv4i_t*)(lds_dst), 0, 0);
#else
    *reinterpret_cast<bf16x8*>(lds_dst) = *reinterpret_cast<const bf16x8*>(gsrc);
#endif
}

__device__ __forceinline__ void wait_async_lds() {
#if HAVE_ASYNC_LDS
    __builtin_amdgcn_s_wait_asynccnt(0);
#endif
}

// ---------------------------------------------------------------------------
// f32 -> bf16 pack (8 elements / thread) -- used for the two weight matrices
// ---------------------------------------------------------------------------
__global__ void __launch_bounds__(256) cvt_f32_bf16(const float* __restrict__ in,
                                                    __bf16* __restrict__ out, int n8) {
    int i = blockIdx.x * 256 + threadIdx.x;
    if (i >= n8) return;
    const float4* p = reinterpret_cast<const float4*>(in) + (size_t)i * 2;
    float4 a = p[0], b = p[1];
    bf16x8 o;
    o[0] = (__bf16)a.x; o[1] = (__bf16)a.y; o[2] = (__bf16)a.z; o[3] = (__bf16)a.w;
    o[4] = (__bf16)b.x; o[5] = (__bf16)b.y; o[6] = (__bf16)b.z; o[7] = (__bf16)b.w;
    reinterpret_cast<bf16x8*>(out)[i] = o;
}

__global__ void __launch_bounds__(256) zero_f32(float* __restrict__ p, int n) {
    int i = blockIdx.x * 256 + threadIdx.x;
    if (i < n) p[i] = 0.0f;
}

// ---------------------------------------------------------------------------
// attn[t][k] = amp_k * exp(-0.5*||x_t - c_k||^2 / s_k^2), normalized over k.
// Fused: also emits x_bf = bf16(x) so x is read from HBM exactly once.
// ---------------------------------------------------------------------------
__global__ void __launch_bounds__(256) attn_kernel(const float* __restrict__ x,
                                                   const float* __restrict__ positions,
                                                   const float* __restrict__ log_scales,
                                                   const float* __restrict__ amplitudes,
                                                   float* __restrict__ attn,
                                                   __bf16* __restrict__ x_bf) {
    __shared__ float pos_lds[KSPL * DDIM];   // 40 KB
    __shared__ float c2_lds[KSPL];
    __shared__ float inv_lds[KSPL];
    __shared__ float amp_lds[KSPL];

    int tid = threadIdx.x;
    for (int c = tid; c < KSPL * DDIM / 4; c += 256)
        async_copy_b128(&pos_lds[c * 4], positions + c * 4);
    wait_async_lds();
    __syncthreads();

    if (tid < KSPL) {
        float s = 0.f;
        for (int d = 0; d < DDIM; ++d) { float p = pos_lds[tid * DDIM + d]; s += p * p; }
        c2_lds[tid] = s;
        float sc = __expf(log_scales[tid]);
        inv_lds[tid] = 0.5f / (sc * sc);
        amp_lds[tid] = amplitudes[tid];
    }
    __syncthreads();

    int wave = tid >> 5, lane = tid & 31;
    int t = blockIdx.x * 8 + wave;
    const float* xr = x + (size_t)t * DDIM;
    __bf16* xb = x_bf + (size_t)t * DDIM;

    float dot[KSPL];
    #pragma unroll
    for (int k = 0; k < KSPL; ++k) dot[k] = 0.f;
    float x2 = 0.f;

    #pragma unroll 4
    for (int it = 0; it < 16; ++it) {
        int d = lane + it * 32;
        float xv = xr[d];
        xb[d] = (__bf16)xv;               // fused conversion (coalesced b16)
        x2 += xv * xv;
        #pragma unroll
        for (int k = 0; k < KSPL; ++k) dot[k] += xv * pos_lds[k * DDIM + d];
    }
    #pragma unroll
    for (int off = 16; off >= 1; off >>= 1) {
        x2 += __shfl_xor(x2, off);
        #pragma unroll
        for (int k = 0; k < KSPL; ++k) dot[k] += __shfl_xor(dot[k], off);
    }

    float g[KSPL], sum = 0.f;
    #pragma unroll
    for (int k = 0; k < KSPL; ++k) {
        float d2 = fmaxf(x2 - 2.f * dot[k] + c2_lds[k], 0.f);
        g[k] = amp_lds[k] * __expf(-d2 * inv_lds[k]);
        sum += g[k];
    }
    float rinv = 1.f / (sum + 1e-8f);
    if (lane == 0) {
        float* ap = attn + (size_t)t * KSPL;
        #pragma unroll
        for (int k = 0; k < KSPL; ++k) ap[k] = g[k] * rinv;
    }
}

// ---------------------------------------------------------------------------
// WMMA fragment builders (ISA 7.12.2, wave32, 16-bit layouts).
// ---------------------------------------------------------------------------
__device__ __forceinline__ v16bf load_a_frag(const __bf16* base /* + d0, row stride 512 */,
                                             int lane) {
    int m = lane & 15, half = lane >> 4;
    const __bf16* rp = base + (size_t)m * DDIM + 8 * half;
    bf16x8 lo = *reinterpret_cast<const bf16x8*>(rp);
    bf16x8 hi = *reinterpret_cast<const bf16x8*>(rp + 16);
    v16bf a;
    #pragma unroll
    for (int i = 0; i < 8; ++i) { a[i] = lo[i]; a[i + 8] = hi[i]; }
    return a;
}

__device__ __forceinline__ v16bf load_b_frag(const __bf16* w /* W + e0*512 + d0 */,
                                             int lane) {
    int n = lane & 15, kg = lane >> 4;
    const __bf16* rp = w + (size_t)n * DDIM + 8 * kg;
    bf16x8 b0 = *reinterpret_cast<const bf16x8*>(rp);
    bf16x8 b1 = *reinterpret_cast<const bf16x8*>(rp + 16);
    v16bf b;
    #pragma unroll
    for (int i = 0; i < 8; ++i) { b[i] = b0[i]; b[i + 8] = b1[i]; }
    return b;
}

// ---------------------------------------------------------------------------
// C = A @ W^T.  Block: 64 rows, A tile 64x512 bf16 (64 KB) async-staged to
// LDS.  Wave: 64(M) x 64(N) macro-tile = 4x4 WMMA accumulators; each B
// fragment is reused by 4 A fragments.
// ---------------------------------------------------------------------------
template <typename OutT>
__device__ __forceinline__ void gemm64_body(const __bf16* __restrict__ A,
                                            const __bf16* __restrict__ W,
                                            OutT* __restrict__ C) {
    __shared__ __bf16 a_lds[64 * DDIM];   // 64 KB
    int tid = threadIdx.x, lane = tid & 31, wave = tid >> 5;
    int s0 = blockIdx.x * 64;
    int n0 = wave * 64;

    // stage A tile: 32768 bf16 = 4096 x 16B chunks, 16 per thread
    const __bf16* Ab = A + (size_t)s0 * DDIM;
    #pragma unroll
    for (int c = 0; c < 16; ++c) {
        int idx = tid + c * 256;
        async_copy_b128(&a_lds[idx * 8], Ab + idx * 8);
    }
    wait_async_lds();
    __syncthreads();

    v8f acc[4][4] = {};   // [mi][j]

    #pragma unroll 4
    for (int kt = 0; kt < 16; ++kt) {
        int d0 = kt * 32;
        v16bf a[4];
        #pragma unroll
        for (int mi = 0; mi < 4; ++mi)
            a[mi] = load_a_frag(&a_lds[mi * 16 * DDIM + d0], lane);
        #pragma unroll
        for (int j = 0; j < 4; ++j) {
            v16bf b = load_b_frag(W + (size_t)(n0 + 16 * j) * DDIM + d0, lane);
            #pragma unroll
            for (int mi = 0; mi < 4; ++mi)
                acc[mi][j] = __builtin_amdgcn_wmma_f32_16x16x32_bf16(
                    false, a[mi], false, b, (short)0, acc[mi][j], false, false);
        }
    }

    int n = lane & 15, half = lane >> 4;
    #pragma unroll
    for (int mi = 0; mi < 4; ++mi)
        #pragma unroll
        for (int j = 0; j < 4; ++j)
            #pragma unroll
            for (int r = 0; r < 8; ++r)
                C[(size_t)(s0 + mi * 16 + r + 8 * half) * DDIM + n0 + 16 * j + n] =
                    (OutT)acc[mi][j][r];
}

__global__ void __launch_bounds__(256) gemm64_bf16(const __bf16* __restrict__ A,
                                                   const __bf16* __restrict__ W,
                                                   __bf16* __restrict__ C) {
    gemm64_body<__bf16>(A, W, C);
}

__global__ void __launch_bounds__(256) gemm64_f32(const __bf16* __restrict__ A,
                                                  const __bf16* __restrict__ W,
                                                  float* __restrict__ C) {
    gemm64_body<float>(A, W, C);
}

// ---------------------------------------------------------------------------
// splat_states[b,k,:] = sum_s attn[b,s,k] * v[b,s,:]   (split-S, f32 atomics)
// Block = (batch, 128-token chunk); thread owns 2 D-columns, 20 K-slots.
// ---------------------------------------------------------------------------
__global__ void __launch_bounds__(256) splat_kernel(const float* __restrict__ attn,
                                                    const __bf16* __restrict__ v,
                                                    float* __restrict__ splat) {
    __shared__ float a_lds[128 * KSPL];   // 10 KB
    int tid = threadIdx.x;
    int chunk = blockIdx.x & 63, b = blockIdx.x >> 6;
    size_t t_base = (size_t)b * SEQ + (size_t)chunk * 128;

    const float* ab = attn + t_base * KSPL;
    for (int c = tid; c < 128 * KSPL / 4; c += 256)   // 640 chunks
        async_copy_b128(&a_lds[c * 4], ab + c * 4);
    wait_async_lds();
    __syncthreads();

    int d0 = tid * 2;
    float accx[KSPL], accy[KSPL];
    #pragma unroll
    for (int k = 0; k < KSPL; ++k) { accx[k] = 0.f; accy[k] = 0.f; }

    for (int s = 0; s < 128; ++s) {
        bf16x2 vv = *reinterpret_cast<const bf16x2*>(v + (t_base + s) * DDIM + d0);
        float v0 = (float)vv[0], v1 = (float)vv[1];
        #pragma unroll
        for (int k = 0; k < KSPL; ++k) {
            float a = a_lds[s * KSPL + k];
            accx[k] += a * v0;
            accy[k] += a * v1;
        }
    }
    float* spb = splat + (size_t)b * KSPL * DDIM;
    #pragma unroll
    for (int k = 0; k < KSPL; ++k) {
        atomicAdd(&spb[k * DDIM + d0], accx[k]);
        atomicAdd(&spb[k * DDIM + d0 + 1], accy[k]);
    }
}

// ---------------------------------------------------------------------------
// mid[t,:] = bf16( sum_k attn[t,k] * splat_states[b,k,:] ).
// Block: 32 tokens; splat (40 KB) + attn tile async-staged to LDS.
// ---------------------------------------------------------------------------
__global__ void __launch_bounds__(256) mid_kernel(const float* __restrict__ attn,
                                                  const float* __restrict__ splat,
                                                  __bf16* __restrict__ mid) {
    __shared__ float splat_lds[KSPL * DDIM];  // 40 KB
    __shared__ float attn_lds[32 * KSPL];     // 2.5 KB

    int tid = threadIdx.x;
    int t0 = blockIdx.x * 32;
    int b = t0 >> 13;
    const float* sp = splat + (size_t)b * KSPL * DDIM;
    const float* ap = attn + (size_t)t0 * KSPL;

    for (int c = tid; c < KSPL * DDIM / 4; c += 256)   // 2560 chunks
        async_copy_b128(&splat_lds[c * 4], sp + c * 4);
    for (int c = tid; c < 32 * KSPL / 4; c += 256)     // 160 chunks
        async_copy_b128(&attn_lds[c * 4], ap + c * 4);
    wait_async_lds();
    __syncthreads();

    int row = tid >> 3;            // 0..31
    int c0 = (tid & 7) * 64;       // 0..448

    float ar[KSPL];
    #pragma unroll
    for (int k = 0; k < KSPL; ++k) ar[k] = attn_lds[row * KSPL + k];

    float4 acc4[16] = {};
    #pragma unroll
    for (int k = 0; k < KSPL; ++k) {
        const float4* srow = reinterpret_cast<const float4*>(&splat_lds[k * DDIM + c0]);
        float ak = ar[k];
        #pragma unroll
        for (int j = 0; j < 16; ++j) {
            float4 s4 = srow[j];
            acc4[j].x += ak * s4.x; acc4[j].y += ak * s4.y;
            acc4[j].z += ak * s4.z; acc4[j].w += ak * s4.w;
        }
    }
    __bf16* mp = mid + (size_t)(t0 + row) * DDIM + c0;
    #pragma unroll
    for (int j = 0; j < 8; ++j) {
        bf16x8 o;
        o[0] = (__bf16)acc4[2 * j].x;     o[1] = (__bf16)acc4[2 * j].y;
        o[2] = (__bf16)acc4[2 * j].z;     o[3] = (__bf16)acc4[2 * j].w;
        o[4] = (__bf16)acc4[2 * j + 1].x; o[5] = (__bf16)acc4[2 * j + 1].y;
        o[6] = (__bf16)acc4[2 * j + 1].z; o[7] = (__bf16)acc4[2 * j + 1].w;
        reinterpret_cast<bf16x8*>(mp)[j] = o;
    }
}

// ---------------------------------------------------------------------------
// Host launcher
// ---------------------------------------------------------------------------
extern "C" void kernel_launch(void* const* d_in, const int* in_sizes, int n_in,
                              void* d_out, int out_size, void* d_ws, size_t ws_size,
                              hipStream_t stream) {
    const float* x          = (const float*)d_in[0];
    const float* positions  = (const float*)d_in[1];
    const float* log_scales = (const float*)d_in[2];
    const float* amplitudes = (const float*)d_in[3];
    const float* w_v        = (const float*)d_in[4];
    const float* w_o        = (const float*)d_in[5];
    float* out = (float*)d_out;

    char* ws = (char*)d_ws;
    size_t off = 0;
    __bf16* x_bf   = (__bf16*)(ws + off); off += (size_t)NTOK * DDIM * 2;     // 33.5 MB
    __bf16* wv_bf  = (__bf16*)(ws + off); off += (size_t)DDIM * DDIM * 2;     // 0.5 MB
    __bf16* wo_bf  = (__bf16*)(ws + off); off += (size_t)DDIM * DDIM * 2;     // 0.5 MB
    __bf16* v_bf   = (__bf16*)(ws + off); off += (size_t)NTOK * DDIM * 2;     // 33.5 MB
    __bf16* mid_bf = (__bf16*)(ws + off); off += (size_t)NTOK * DDIM * 2;     // 33.5 MB
    float*  attn   = (float*)(ws + off);  off += (size_t)NTOK * KSPL * 4;     // 2.6 MB
    float*  splat  = (float*)(ws + off);  off += (size_t)4 * KSPL * DDIM * 4; // 160 KB

    cvt_f32_bf16<<<(DDIM * DDIM / 8 + 255) / 256, 256, 0, stream>>>(w_v, wv_bf, DDIM * DDIM / 8);
    cvt_f32_bf16<<<(DDIM * DDIM / 8 + 255) / 256, 256, 0, stream>>>(w_o, wo_bf, DDIM * DDIM / 8);
    zero_f32<<<(4 * KSPL * DDIM + 255) / 256, 256, 0, stream>>>(splat, 4 * KSPL * DDIM);

    // attn + fused x -> bf16 conversion (single HBM pass over x)
    attn_kernel<<<NTOK / 8, 256, 0, stream>>>(x, positions, log_scales, amplitudes,
                                              attn, x_bf);
    gemm64_bf16<<<NTOK / 64, 256, 0, stream>>>(x_bf, wv_bf, v_bf);
    splat_kernel<<<NTOK / 128, 256, 0, stream>>>(attn, v_bf, splat);
    mid_kernel<<<NTOK / 32, 256, 0, stream>>>(attn, splat, mid_bf);
    gemm64_f32<<<NTOK / 64, 256, 0, stream>>>(mid_bf, wo_bf, out);
}
